// DecoderRNN_78176994722555
// MI455X (gfx1250) — matrix operand order repaired and checked by reading, compiled, not run on hardware
//
#include <hip/hip_runtime.h>
#include <hip/hip_bf16.h>

typedef __attribute__((ext_vector_type(16))) __bf16 v16bf;
typedef __attribute__((ext_vector_type(8)))  __bf16 v8bf;
typedef __attribute__((ext_vector_type(8)))  float  v8f;
typedef __attribute__((ext_vector_type(4)))  unsigned int u32x4;
typedef __attribute__((ext_vector_type(8)))  int i32x8;
typedef __attribute__((ext_vector_type(4)))  int i32x4;

#define B_  64
#define S_  32
#define P_  49
#define F_  2048
#define E_  512
#define H_  512
#define V_  32000
#define KCAT 3072     // E + F + H

#define KC     512    // k-chunk staged in LDS per TDM transfer
#define KCPAD  520    // row stride in LDS elements (1024B + 16B TDM pad)

static __device__ __forceinline__ __bf16 f2bf(float f) {
    union { float f; unsigned u; } v; v.f = f;
    unsigned r = v.u + 0x7FFFu + ((v.u >> 16) & 1u);   // round-to-nearest-even
    unsigned short h = (unsigned short)(r >> 16);
    __bf16 out;
    __builtin_memcpy(&out, &h, 2);
    return out;
}

// ---------------------------------------------------------------------------
// C[M,N] = A[M,K](bf16) x W[N,K]^T(bf16) + bias(f32), f32 accum.
// Block: 256 threads = 8 waves as 2(m) x 4(n); block tile 32x256.
// A tile (32 x KC) staged in LDS by the Tensor Data Mover per k-chunk,
// with TDM padding for conflict-free ds_load_b128 fragment reads.
// B (weights) streams from global/L2 per wave.
// Requires M%32==0, N%256==0, K%512==0.
// ---------------------------------------------------------------------------
__global__ void wmma_gemm_bf16(const __bf16* __restrict__ A, int lda,
                               const __bf16* __restrict__ W, int ldw,
                               const float* __restrict__ bias,
                               float* __restrict__ C, long long ldc,
                               int M, int N, int K) {
    __shared__ __bf16 sA[32 * KCPAD];

    const int lane = threadIdx.x & 31;
    const int wave = threadIdx.x >> 5;
    const int mt = wave >> 2;                 // 0..1 : m sub-tile in block
    const int ns = wave & 3;                  // 0..3 : 64-wide n strip in block
    const int bn = blockIdx.x;                // N/256 blocks
    const int bm = blockIdx.y;                // M/32  blocks

    const int koffA = (lane & 16) ? 8  : 0;   // ISA 16-bit A fragment layout
    const int koffB = (lane & 16) ? 8  : 0;   // ISA 16-bit B fragment layout

    const int nrow0 = (bn << 8) + (ns << 6) + (lane & 15);
    const __bf16* __restrict__ brow = W + (size_t)nrow0 * ldw + koffB;

    // LDS fragment base for this lane (padded row stride)
    const __bf16* sArow = sA + (size_t)((mt << 4) + (lane & 15)) * KCPAD + koffA;

    const unsigned lds_base = (unsigned)(unsigned long long)(const void*)sA;

    v8f acc[4] = {};
    const int nchunks = K / KC;
    for (int kc = 0; kc < nchunks; ++kc) {
        __syncthreads();                      // previous chunk fully consumed
        if (wave == 0) {
            const unsigned long long gaddr =
                (unsigned long long)(const void*)(A + (size_t)(bm << 5) * lda
                                                    + (size_t)kc * KC);
#if __has_builtin(__builtin_amdgcn_tensor_load_to_lds)
            // ---- Tensor DMA descriptor (D#), 2D tile 512 x 32, bf16 ----
            u32x4 g0;
            g0[0] = 1u;                                      // count=1, no gather
            g0[1] = lds_base;                                // lds_addr
            g0[2] = (unsigned)gaddr;                         // global_addr[31:0]
            g0[3] = ((unsigned)(gaddr >> 32) & 0x01FFFFFFu)  // global_addr[56:32]
                    | 0x80000000u;                           // type=2 (image)
            i32x8 g1;
            g1[0] = (int)((1u << 16)       // data_size = 2B
                        | (1u << 20)       // pad_enable
                        | (7u << 22)       // pad_interval = 256 dwords (1024B)
                        | (3u << 25));     // pad_amount  = 4 dwords (16B)
            g1[1] = (int)((unsigned)(KC & 0xFFFF) << 16);    // tensor_dim0[15:0]
            g1[2] = (int)((KC >> 16) | (32u << 16));         // t_dim0[31:16] | tensor_dim1
            g1[3] = (int)((unsigned)KC << 16);               // tile_dim0
            g1[4] = 32;                                      // tile_dim1 (tile_dim2=0)
            g1[5] = lda;                                     // tensor_dim0_stride[31:0]
            g1[6] = 0;
            g1[7] = 0;
            const i32x4 gz4 = {0, 0, 0, 0};
            const i32x8 gz8 = {0, 0, 0, 0, 0, 0, 0, 0};
            __builtin_amdgcn_tensor_load_to_lds(g0, g1, gz4, gz4, gz8, 0);
            __builtin_amdgcn_s_wait_tensorcnt(0);
#else
            // Fallback: cooperative copy into the same padded layout (wave 0).
            for (int e = lane; e < 32 * KC; e += 32) {
                const int r = e >> 9, cc = e & (KC - 1);
                sA[r * KCPAD + cc] = A[(size_t)((bm << 5) + r) * lda
                                       + (size_t)kc * KC + cc];
            }
            (void)gaddr;
#endif
        }
        __syncthreads();                      // staged tile visible to all waves

        const int kbase = kc * KC;
        for (int k0 = 0; k0 < KC; k0 += 32) {
            // Issue all 8 B loads before any WMMA (partial-wait friendly).
            v8bf bl[4], bh[4];
#pragma unroll
            for (int j = 0; j < 4; ++j) {
                const __bf16* bp = brow + (size_t)(j << 4) * ldw + kbase + k0;
                bl[j] = *(const v8bf*)bp;
                bh[j] = *(const v8bf*)(bp + 16);
            }
            const v8bf alo = *(const v8bf*)(sArow + k0);        // ds_load_b128
            const v8bf ahi = *(const v8bf*)(sArow + k0 + 16);   // ds_load_b128
            const v16bf a = __builtin_shufflevector(alo, ahi,
                0,1,2,3,4,5,6,7,8,9,10,11,12,13,14,15);
#pragma unroll
            for (int j = 0; j < 4; ++j) {
                const v16bf b = __builtin_shufflevector(bl[j], bh[j],
                    0,1,2,3,4,5,6,7,8,9,10,11,12,13,14,15);
                acc[j] = __builtin_amdgcn_wmma_f32_16x16x32_bf16(
                             false, a, false, b, (short)0, acc[j], false, false);
            }
        }
    }

    const int mbase = (bm << 5) + (mt << 4) + ((lane & 16) ? 8 : 0);
#pragma unroll
    for (int j = 0; j < 4; ++j) {
        const int ncol = (bn << 8) + (ns << 6) + (j << 4) + (lane & 15);
        const float bv = bias ? bias[ncol] : 0.0f;
#pragma unroll
        for (int r = 0; r < 8; ++r)
            C[(size_t)(mbase + r) * (size_t)ldc + ncol] = acc[j][r] + bv;
    }
}

// --------------------------- support kernels -------------------------------

__global__ void cvt_bf16_kernel(const float* __restrict__ src,
                                __bf16* __restrict__ dst, long long n) {
    long long i = ((long long)blockIdx.x * blockDim.x + threadIdx.x) * 4;
    if (i + 3 < n) {
        const float4 v = *(const float4*)(src + i);
        dst[i]     = f2bf(v.x);
        dst[i + 1] = f2bf(v.y);
        dst[i + 2] = f2bf(v.z);
        dst[i + 3] = f2bf(v.w);
    } else {
        for (; i < n; ++i) dst[i] = f2bf(src[i]);
    }
}

__global__ void embed_kernel(const int* __restrict__ cap,
                             const float* __restrict__ table,
                             float* __restrict__ emb) {
    const int bs = blockIdx.x;                 // B*S blocks
    const int tok = cap[bs];
    const float* src = table + (size_t)tok * E_;
    float* dst = emb + (size_t)bs * E_;
    for (int i = threadIdx.x; i < E_; i += blockDim.x) dst[i] = src[i];
}

__global__ void meanfeat_kernel(const float* __restrict__ feat,
                                __bf16* __restrict__ mf) {
    const int idx = blockIdx.x * blockDim.x + threadIdx.x;     // B*F
    if (idx >= B_ * F_) return;
    const int b = idx >> 11, f = idx & (F_ - 1);
    const float* p = feat + ((size_t)b * P_) * F_ + f;
    float s = 0.f;
    for (int j = 0; j < P_; ++j) s += p[(size_t)j * F_];
    mf[idx] = f2bf(s * (1.0f / 49.0f));
}

__global__ void concat_w_kernel(const float* __restrict__ Wih,
                                const float* __restrict__ Whh,
                                const float* __restrict__ bih,
                                const float* __restrict__ bhh,
                                __bf16* __restrict__ Wcat,
                                float* __restrict__ bcat) {
    const int idx = blockIdx.x * blockDim.x + threadIdx.x;
    const int total = (4 * H_) * KCAT;
    if (idx < total) {
        const int n = idx / KCAT, k = idx - n * KCAT;
        const float w = (k < E_ + F_) ? Wih[(size_t)n * (E_ + F_) + k]
                                      : Whh[(size_t)n * H_ + (k - (E_ + F_))];
        Wcat[idx] = f2bf(w);
    }
    if (idx < 4 * H_) bcat[idx] = bih[idx] + bhh[idx];
}

// One block per batch element: att2 -> tanh-score -> softmax -> context.
__global__ void attention_kernel(const float* __restrict__ h,
                                 const float* __restrict__ att1,
                                 const float* __restrict__ Ua,
                                 const float* __restrict__ bUa,
                                 const float* __restrict__ va,
                                 const float* __restrict__ bva,
                                 const float* __restrict__ feat,
                                 float* __restrict__ attw_out,
                                 float* __restrict__ context,
                                 int s) {
    __shared__ float sh[H_];
    __shared__ float satt2[H_];
    __shared__ float ssc[64];
    const int b = blockIdx.x;

    for (int i = threadIdx.x; i < H_; i += blockDim.x) sh[i] = h[(size_t)b * H_ + i];
    __syncthreads();

    for (int j = threadIdx.x; j < H_; j += blockDim.x) {
        const float* u = Ua + (size_t)j * H_;
        float acc = bUa[j];
        for (int k = 0; k < H_; ++k) acc += sh[k] * u[k];
        satt2[j] = acc;
    }
    __syncthreads();

    if (threadIdx.x < P_) {
        const float* ap = att1 + ((size_t)b * P_ + threadIdx.x) * H_;
        float acc = bva[0];
        for (int j = 0; j < H_; ++j) acc += va[j] * tanhf(ap[j] + satt2[j]);
        ssc[threadIdx.x] = acc;
    }
    __syncthreads();

    if (threadIdx.x == 0) {
        float mx = ssc[0];
        for (int p = 1; p < P_; ++p) mx = fmaxf(mx, ssc[p]);
        float sum = 0.f;
        for (int p = 0; p < P_; ++p) { float e = __expf(ssc[p] - mx); ssc[p] = e; sum += e; }
        const float inv = 1.0f / sum;
        for (int p = 0; p < P_; ++p) ssc[p] *= inv;
    }
    __syncthreads();

    if (threadIdx.x < P_)
        attw_out[((size_t)b * S_ + s) * P_ + threadIdx.x] = ssc[threadIdx.x];

    const float* fb = feat + (size_t)b * P_ * F_;
    for (int f = threadIdx.x; f < F_; f += blockDim.x) {
        float acc = 0.f;
        for (int p = 0; p < P_; ++p) acc += ssc[p] * fb[(size_t)p * F_ + f];
        context[(size_t)b * F_ + f] = acc;
    }
}

__global__ void build_xh_kernel(const float* __restrict__ emb,
                                const float* __restrict__ context,
                                const float* __restrict__ h,
                                __bf16* __restrict__ xh, int s) {
    const int idx = blockIdx.x * blockDim.x + threadIdx.x;     // B*KCAT
    if (idx >= B_ * KCAT) return;
    const int b = idx / KCAT, k = idx - b * KCAT;
    float v;
    if (k < E_)            v = emb[((size_t)b * S_ + s) * E_ + k];
    else if (k < E_ + F_)  v = context[(size_t)b * F_ + (k - E_)];
    else                   v = h[(size_t)b * H_ + (k - E_ - F_)];
    xh[idx] = f2bf(v);
}

__global__ void lstm_kernel(const float* __restrict__ gates,
                            float* __restrict__ h, float* __restrict__ c,
                            __bf16* __restrict__ Hs, int s) {
    const int idx = blockIdx.x * blockDim.x + threadIdx.x;     // B*H
    if (idx >= B_ * H_) return;
    const int b = idx >> 9, j = idx & (H_ - 1);
    const float* g = gates + (size_t)b * (4 * H_);
    const float gi = g[j], gf = g[H_ + j], gg = g[2 * H_ + j], go = g[3 * H_ + j];
    const float si = 1.f / (1.f + __expf(-gi));
    const float sf = 1.f / (1.f + __expf(-gf));
    const float so = 1.f / (1.f + __expf(-go));
    const float cn = sf * c[idx] + si * tanhf(gg);
    const float hn = so * tanhf(cn);
    c[idx] = cn;
    h[idx] = hn;
    Hs[((size_t)b * S_ + s) * H_ + j] = f2bf(hn);   // row (b*S+s) for final fc GEMM
}

// ---------------------------------------------------------------------------

static inline dim3 gemm_grid(int M, int N) { return dim3(N / 256, M / 32); }

static inline int cvt_blocks(long long n) {
    return (int)((n / 4 + 255) / 256);
}

extern "C" void kernel_launch(void* const* d_in, const int* in_sizes, int n_in,
                              void* d_out, int out_size, void* d_ws, size_t ws_size,
                              hipStream_t stream) {
    const int*   captions = (const int*)  d_in[0];
    const float* features = (const float*)d_in[1];
    const float* table    = (const float*)d_in[2];
    const float* W_a      = (const float*)d_in[3];
    const float* b_Wa     = (const float*)d_in[4];
    const float* U_a      = (const float*)d_in[5];
    const float* b_Ua     = (const float*)d_in[6];
    const float* v_a      = (const float*)d_in[7];
    const float* b_va     = (const float*)d_in[8];
    const float* W_ih     = (const float*)d_in[9];
    const float* b_ih     = (const float*)d_in[10];
    const float* W_hh     = (const float*)d_in[11];
    const float* b_hh     = (const float*)d_in[12];
    const float* fc_W     = (const float*)d_in[13];
    const float* fc_b     = (const float*)d_in[14];
    const float* Wh       = (const float*)d_in[15];
    const float* bh       = (const float*)d_in[16];
    const float* Wc       = (const float*)d_in[17];
    const float* bc       = (const float*)d_in[18];

    float* out_logits = (float*)d_out;                                // [B,S,V]
    float* out_attw   = out_logits + (size_t)B_ * S_ * V_;            // [B,S,P]

    // ---- workspace carve-up: fp32 regions first, then bf16 regions ----
    float* wf = (float*)d_ws;
    float* Emb     = wf;  wf += (size_t)B_ * S_ * E_;
    float* att1    = wf;  wf += (size_t)B_ * P_ * H_;
    float* bcat    = wf;  wf += 4 * H_;
    float* hbuf    = wf;  wf += (size_t)B_ * H_;
    float* cbuf    = wf;  wf += (size_t)B_ * H_;
    float* gates   = wf;  wf += (size_t)B_ * (4 * H_);
    float* context = wf;  wf += (size_t)B_ * F_;

    __bf16* wb = (__bf16*)wf;
    __bf16* meanf   = wb;  wb += (size_t)B_ * F_;
    __bf16* Wcat    = wb;  wb += (size_t)(4 * H_) * KCAT;
    __bf16* xh      = wb;  wb += (size_t)B_ * KCAT;
    __bf16* Hs      = wb;  wb += (size_t)B_ * S_ * H_;
    __bf16* feat_bf = wb;  wb += (size_t)B_ * P_ * F_;
    __bf16* Wa_bf   = wb;  wb += (size_t)H_ * F_;
    __bf16* Wh_bf   = wb;  wb += (size_t)H_ * F_;
    __bf16* Wc_bf   = wb;  wb += (size_t)H_ * F_;
    __bf16* fcW_bf  = wb;  wb += (size_t)V_ * H_;

    // ---- one-time precompute / precision conversion ----
    embed_kernel<<<B_ * S_, 256, 0, stream>>>(captions, table, Emb);
    meanfeat_kernel<<<(B_ * F_ + 255) / 256, 256, 0, stream>>>(features, meanf);
    concat_w_kernel<<<((4 * H_) * KCAT + 255) / 256, 256, 0, stream>>>(
        W_ih, W_hh, b_ih, b_hh, Wcat, bcat);

    const long long n_feat = (long long)B_ * P_ * F_;
    const long long n_hf   = (long long)H_ * F_;
    const long long n_fc   = (long long)V_ * H_;
    cvt_bf16_kernel<<<cvt_blocks(n_feat), 256, 0, stream>>>(features, feat_bf, n_feat);
    cvt_bf16_kernel<<<cvt_blocks(n_hf),   256, 0, stream>>>(W_a,  Wa_bf, n_hf);
    cvt_bf16_kernel<<<cvt_blocks(n_hf),   256, 0, stream>>>(Wh,   Wh_bf, n_hf);
    cvt_bf16_kernel<<<cvt_blocks(n_hf),   256, 0, stream>>>(Wc,   Wc_bf, n_hf);
    cvt_bf16_kernel<<<cvt_blocks(n_fc),   256, 0, stream>>>(fc_W, fcW_bf, n_fc);

    // h0 = meanf @ Wh^T + bh ; c0 = meanf @ Wc^T + bc   (M=64,N=512,K=2048)
    wmma_gemm_bf16<<<gemm_grid(B_, H_), 256, 0, stream>>>(
        meanf, F_, Wh_bf, F_, bh, hbuf, H_, B_, H_, F_);
    wmma_gemm_bf16<<<gemm_grid(B_, H_), 256, 0, stream>>>(
        meanf, F_, Wc_bf, F_, bc, cbuf, H_, B_, H_, F_);

    // att1 = features @ W_a^T + b_Wa   (M=3136,N=512,K=2048)
    wmma_gemm_bf16<<<gemm_grid(B_ * P_, H_), 256, 0, stream>>>(
        feat_bf, F_, Wa_bf, F_, b_Wa, att1, H_, B_ * P_, H_, F_);

    // ---- recurrence ----
    for (int s = 0; s < S_; ++s) {
        attention_kernel<<<B_, 256, 0, stream>>>(
            hbuf, att1, U_a, b_Ua, v_a, b_va, features, out_attw, context, s);
        build_xh_kernel<<<(B_ * KCAT + 255) / 256, 256, 0, stream>>>(
            Emb, context, hbuf, xh, s);
        // gates = xh @ Wcat^T + (b_ih+b_hh)   (M=64,N=2048,K=3072)
        wmma_gemm_bf16<<<gemm_grid(B_, 4 * H_), 256, 0, stream>>>(
            xh, KCAT, Wcat, KCAT, bcat, gates, 4 * H_, B_, 4 * H_, KCAT);
        lstm_kernel<<<(B_ * H_ + 255) / 256, 256, 0, stream>>>(
            gates, hbuf, cbuf, Hs, s);
    }

    // ---- hoisted vocabulary projection: logits = Hs @ fc_W^T + fc_b ----
    // (M=2048, N=32000, K=512) — dominant GEMM, fully parallel, L2-resident weights.
    wmma_gemm_bf16<<<gemm_grid(B_ * S_, V_), 256, 0, stream>>>(
        Hs, H_, fcW_bf, H_, fc_b, out_logits, V_, B_ * S_, V_, H_);
}
